// LlamaRegressor_44976897524258
// MI455X (gfx1250) — matrix operand order repaired and checked by reading, not compiled, not run on hardware
//
#include <hip/hip_runtime.h>

// TinyLlama-ish forward (L=4, D=2048, S=2048, FF=5632, H=32, KV=4, HD=64)
// CDNA5 / gfx1250: wave32, bf16 WMMA 16x16x32 with f32 accumulation.
// Weights are int8-valued int32; dequantized to bf16 (exact) right before each
// GEMM so the WMMA inner loop carries no conversion VALU work.

#define DD   2048
#define SS   2048
#define FFD  5632
#define NH   32
#define NKV  4
#define HD64 64

typedef __attribute__((ext_vector_type(16))) __bf16 v16bf;
typedef __attribute__((ext_vector_type(8)))  __bf16 v8bf;
typedef __attribute__((ext_vector_type(4)))  __bf16 v4bf;
typedef __attribute__((ext_vector_type(8)))  float  v8f;
typedef __attribute__((ext_vector_type(4)))  int    v4i;

union U16 { v16bf v; v8bf h[2]; };
union UI4 { v4i v; int s[4]; };
union UB4 { v4bf v; __bf16 s[4]; };

static __device__ __forceinline__ v8f wmma_bf16(v16bf a, v16bf b, v8f c) {
  // D = A(16x32 bf16) * B(32x16 bf16) + C(16x16 f32)
  return __builtin_amdgcn_wmma_f32_16x16x32_bf16(false, a, false, b, (short)0, c, false, false);
}

// ---------------------------------------------------------------------------
// Weight dequant: int32 (|w|<=127, exact in bf16) -> bf16. n multiple of 4.
// ---------------------------------------------------------------------------
__global__ void __launch_bounds__(256)
dequant_kernel(const int* __restrict__ W, __bf16* __restrict__ out, int n) {
  int i = (int)(blockIdx.x * 256 + threadIdx.x) * 4;
  if (i >= n) return;
  UI4 u; u.v = *(const v4i*)(W + i);
  UB4 b;
#pragma unroll
  for (int e = 0; e < 4; ++e) b.s[e] = (__bf16)(float)u.s[e];
  *(v4bf*)(out + i) = b.v;
}

// ---------------------------------------------------------------------------
// GEMM: out[M,N] = act[M,K](bf16) x Wb[N,K](bf16) * scale[N]
// mode 0: outF[M,N] += ...   (fp32 residual accumulate)
// mode 1: outB[M,N]  = bf16(...)
// mode 2: outB[N,M]  = bf16(...)   (transposed store, used for V)
// One wave computes a 32(M) x 64(N) tile; K-step 32 => 8 WMMA / 12 b128 loads.
// ---------------------------------------------------------------------------
__global__ void __launch_bounds__(256)
gemm_bf16w(const __bf16* __restrict__ A, const __bf16* __restrict__ W,
           const float* __restrict__ scale, int M, int N, int K,
           int mode, float* __restrict__ outF, __bf16* __restrict__ outB) {
  int gw = (int)((blockIdx.x * blockDim.x + threadIdx.x) >> 5);
  int nT = N >> 6, mT = M >> 5;
  if (gw >= mT * nT) return;           // wave-uniform exit
  int mt = gw / nT, nb = gw - mt * nT;
  int m0 = mt << 5, n0 = nb << 6;
  int lane = (int)(threadIdx.x & 31);
  int g = lane >> 4, r = lane & 15;

  v8f c[2][4];
#pragma unroll
  for (int p = 0; p < 2; ++p)
#pragma unroll
    for (int t = 0; t < 4; ++t) c[p][t] = (v8f){};

  const __bf16* a0 = A + (size_t)(m0 + r) * K;        // rows m0+r, m0+16+r
  const __bf16* a1 = A + (size_t)(m0 + 16 + r) * K;

  for (int k = 0; k < K; k += 32) {
    U16 af0, af1;   // A frag: e<8 -> k+g*8+e ; e>=8 -> k+16+g*8+(e-8)
    af0.h[0] = *(const v8bf*)(a0 + k + g * 8);
    af0.h[1] = *(const v8bf*)(a0 + k + 16 + g * 8);
    af1.h[0] = *(const v8bf*)(a1 + k + g * 8);
    af1.h[1] = *(const v8bf*)(a1 + k + 16 + g * 8);
    if (k + 32 < K) __builtin_prefetch(a0 + k + 32, 0, 0);
#pragma unroll
    for (int t = 0; t < 4; ++t) {      // B frag: e<8 -> k+g*8+e ; e>=8 -> k+16+g*8+(e-8)
      const __bf16* wr = W + (size_t)(n0 + t * 16 + r) * K + k;
      U16 bf; bf.h[0] = *(const v8bf*)(wr + g * 8); bf.h[1] = *(const v8bf*)(wr + 16 + g * 8);
      c[0][t] = wmma_bf16(af0.v, bf.v, c[0][t]);
      c[1][t] = wmma_bf16(af1.v, bf.v, c[1][t]);
    }
  }

#pragma unroll
  for (int t = 0; t < 4; ++t) {
    int col = n0 + t * 16 + r;
    float sc = scale[col];
#pragma unroll
    for (int p = 0; p < 2; ++p)
#pragma unroll
      for (int i = 0; i < 8; ++i) {   // C layout: reg i -> row i+8*g, col = lane&15
        int row = m0 + p * 16 + i + 8 * g;
        float v = c[p][t][i] * sc;
        if (mode == 0)      outF[(size_t)row * N + col] += v;
        else if (mode == 1) outB[(size_t)row * N + col] = (__bf16)v;
        else                outB[(size_t)col * M + row] = (__bf16)v;
      }
  }
}

// ---------------------------------------------------------------------------
// Flash attention, GQA. One wave per (head, 16-query tile).
// Q:[S, H*64] bf16 (roped), Kb:[S, KV*64] bf16 (roped), Vt:[KV*64, S] bf16.
// Out: O:[S, H*64] bf16.
// ---------------------------------------------------------------------------
__global__ void __launch_bounds__(256)
attn_kernel(const __bf16* __restrict__ Q, const __bf16* __restrict__ Kb,
            const __bf16* __restrict__ Vt, __bf16* __restrict__ O) {
  __shared__ __align__(16) __bf16 pl[8][16][32];   // per-wave P tile staging
  int wave = (int)(threadIdx.x >> 5), lane = (int)(threadIdx.x & 31);
  int g = lane >> 4, r = lane & 15;
  int head = (int)blockIdx.x;
  int q0 = ((int)blockIdx.y * 8 + wave) * 16;
  int kvh = head >> 3;                              // 8 Q heads per KV head

  // Q fragments for hd 0..31 and 32..63
  U16 alo, ahi;
  const __bf16* qrow = Q + (size_t)(q0 + r) * DD + head * HD64;
  alo.h[0] = *(const v8bf*)(qrow + g * 8);
  alo.h[1] = *(const v8bf*)(qrow + 16 + g * 8);
  ahi.h[0] = *(const v8bf*)(qrow + 32 + g * 8);
  ahi.h[1] = *(const v8bf*)(qrow + 48 + g * 8);

  float m[8], l[8], fac[8];
  v8f o[4];
#pragma unroll
  for (int i = 0; i < 8; ++i) { m[i] = -1e30f; l[i] = 0.f; }
#pragma unroll
  for (int t = 0; t < 4; ++t) o[t] = (v8f){};

  for (int k0 = 0; k0 < q0 + 16; k0 += 32) {
    v8f s[2];
#pragma unroll
    for (int j = 0; j < 2; ++j) {                  // two 16-key score tiles
      int kb = k0 + 16 * j;
      const __bf16* krow = Kb + (size_t)(kb + r) * (NKV * HD64) + kvh * HD64;
      U16 blo, bhi;                                // B frag: e<8 -> hd = base + g*8 + e ; e>=8 -> base + 16 + g*8 + (e-8)
      blo.h[0] = *(const v8bf*)(krow + g * 8);
      blo.h[1] = *(const v8bf*)(krow + 16 + g * 8);
      bhi.h[0] = *(const v8bf*)(krow + 32 + g * 8);
      bhi.h[1] = *(const v8bf*)(krow + 48 + g * 8);
      v8f cc = {};
      cc = wmma_bf16(alo.v, blo.v, cc);
      cc = wmma_bf16(ahi.v, bhi.v, cc);
      s[j] = cc;
    }
    // online softmax per row (row of reg i = q0 + i + 8*g; col = lane&15)
#pragma unroll
    for (int i = 0; i < 8; ++i) {
      int qg = q0 + i + 8 * g;
      float v0 = s[0][i] * 0.125f; if (k0 + r      > qg) v0 = -1e30f;
      float v1 = s[1][i] * 0.125f; if (k0 + 16 + r > qg) v1 = -1e30f;
      float mx = fmaxf(v0, v1);
#pragma unroll
      for (int off = 1; off < 16; off <<= 1) mx = fmaxf(mx, __shfl_xor(mx, off));
      float mn = fmaxf(m[i], mx);
      float f = __expf(m[i] - mn);
      m[i] = mn;
      float p0 = __expf(v0 - mn), p1 = __expf(v1 - mn);
      float rs = p0 + p1;
#pragma unroll
      for (int off = 1; off < 16; off <<= 1) rs += __shfl_xor(rs, off);
      l[i] = l[i] * f + rs;
      fac[i] = f;
      pl[wave][i + 8 * g][r]      = (__bf16)p0;
      pl[wave][i + 8 * g][16 + r] = (__bf16)p1;
    }
    asm volatile("s_wait_dscnt 0" ::: "memory");   // LDS in-order + block reordering
    U16 pf;                                        // P as A-frag [16q x 32key]
    pf.h[0] = *(const v8bf*)&pl[wave][r][g * 8];
    pf.h[1] = *(const v8bf*)&pl[wave][r][16 + g * 8];
#pragma unroll
    for (int t = 0; t < 4; ++t) {                  // P @ V over 32 keys, 4 hd subtiles
      const __bf16* vrow = Vt + (size_t)(kvh * HD64 + t * 16 + r) * SS + k0;
      U16 bf; bf.h[0] = *(const v8bf*)(vrow + g * 8); bf.h[1] = *(const v8bf*)(vrow + 16 + g * 8);
      v8f ot = o[t];
#pragma unroll
      for (int i = 0; i < 8; ++i) ot[i] *= fac[i];
      o[t] = wmma_bf16(pf.v, bf.v, ot);
    }
  }
#pragma unroll
  for (int t = 0; t < 4; ++t)
#pragma unroll
    for (int i = 0; i < 8; ++i) {
      float v = o[t][i] / l[i];
      O[(size_t)(q0 + i + 8 * g) * DD + head * HD64 + t * 16 + r] = (__bf16)v;
    }
}

// ---------------------------------------------------------------------------
__global__ void __launch_bounds__(256)
embed_kernel(const int* __restrict__ ids, const float* __restrict__ emb,
             float* __restrict__ x) {
  int row = (int)blockIdx.x;
  int id = ids[row];
  for (int j = (int)threadIdx.x; j < DD; j += 256)
    x[(size_t)row * DD + j] = emb[(size_t)id * DD + j];
}

__global__ void __launch_bounds__(256)
rms_kernel(const float* __restrict__ x, const float* __restrict__ w,
           __bf16* __restrict__ out) {
  __shared__ float red[256];
  int row = (int)blockIdx.x;
  const float* xr = x + (size_t)row * DD;
  float s = 0.f;
  for (int j = (int)threadIdx.x; j < DD; j += 256) s += xr[j] * xr[j];
  red[threadIdx.x] = s;
  __syncthreads();
  for (int st = 128; st > 0; st >>= 1) {
    if ((int)threadIdx.x < st) red[threadIdx.x] += red[threadIdx.x + st];
    __syncthreads();
  }
  float rs = rsqrtf(red[0] / (float)DD + 1e-5f);
  for (int j = (int)threadIdx.x; j < DD; j += 256)
    out[(size_t)row * DD + j] = (__bf16)(xr[j] * rs * w[j]);
}

__global__ void __launch_bounds__(256)
rope_kernel(__bf16* __restrict__ q, __bf16* __restrict__ k, int n) {
  int idx = (int)(blockIdx.x * 256 + threadIdx.x);
  if (idx >= n) return;
  int d = idx & 31;
  int rest = idx >> 5;
  int hh = rest % (NH + NKV);
  int s = rest / (NH + NKV);
  __bf16* p = (hh < NH) ? (q + (size_t)s * DD + hh * HD64)
                        : (k + (size_t)s * (NKV * HD64) + (hh - NH) * HD64);
  float fr = __powf(10000.f, -(float)(2 * d) * (1.f / 64.f));
  float ang = (float)s * fr;
  float sn, cs;
  sincosf(ang, &sn, &cs);
  float x0 = (float)p[d], x1 = (float)p[d + 32];
  p[d]      = (__bf16)(x0 * cs - x1 * sn);
  p[d + 32] = (__bf16)(x1 * cs + x0 * sn);
}

__global__ void __launch_bounds__(256)
silumul_kernel(const __bf16* __restrict__ gb, const __bf16* __restrict__ ub,
               __bf16* __restrict__ out, int n) {
  int i = (int)(blockIdx.x * 256 + threadIdx.x);
  if (i >= n) return;
  float gv = (float)gb[i], uv = (float)ub[i];
  float s = gv / (1.f + __expf(-gv));
  out[i] = (__bf16)(s * uv);
}

__global__ void __launch_bounds__(512)
head_kernel(const float* __restrict__ x, const float* __restrict__ fn,
            const float* __restrict__ w1, const float* __restrict__ b1,
            const float* __restrict__ w2, const float* __restrict__ b2,
            float* __restrict__ out) {
  __shared__ float red[512];
  __shared__ float hmid[512];
  int t = (int)threadIdx.x;
  const float* xr = x + (size_t)(SS - 1) * DD;
  float s = 0.f;
  for (int j = t; j < DD; j += 512) s += xr[j] * xr[j];
  red[t] = s;
  __syncthreads();
  for (int st = 256; st > 0; st >>= 1) {
    if (t < st) red[t] += red[t + st];
    __syncthreads();
  }
  float rs = rsqrtf(red[0] / (float)DD + 1e-5f);
  float acc = b1[t];
  for (int j = 0; j < DD; ++j) acc += (xr[j] * rs * fn[j]) * w1[(size_t)t * DD + j];
  hmid[t] = fmaxf(acc, 0.f);
  __syncthreads();
  if (t < 3) {
    float a = b2[t];
    for (int j = 0; j < 512; ++j) a += hmid[j] * w2[t * 512 + j];
    out[t] = a;
  }
}

// ---------------------------------------------------------------------------
extern "C" void kernel_launch(void* const* d_in, const int* in_sizes, int n_in,
                              void* d_out, int out_size, void* d_ws, size_t ws_size,
                              hipStream_t stream) {
  (void)in_sizes; (void)n_in; (void)out_size; (void)ws_size;
  const int*   ids   = (const int*)d_in[0];
  const float* emb   = (const float*)d_in[1];
  const int*   Wq    = (const int*)d_in[2];
  const float* sq    = (const float*)d_in[3];
  const int*   Wk    = (const int*)d_in[4];
  const float* sk    = (const float*)d_in[5];
  const int*   Wv    = (const int*)d_in[6];
  const float* sv    = (const float*)d_in[7];
  const int*   Wo    = (const int*)d_in[8];
  const float* so    = (const float*)d_in[9];
  const int*   Wg    = (const int*)d_in[10];
  const float* sg    = (const float*)d_in[11];
  const int*   Wu    = (const int*)d_in[12];
  const float* su    = (const float*)d_in[13];
  const int*   Wd    = (const int*)d_in[14];
  const float* sd    = (const float*)d_in[15];
  const float* lnA   = (const float*)d_in[16];
  const float* lnM   = (const float*)d_in[17];
  const float* fnorm = (const float*)d_in[18];
  const float* w1    = (const float*)d_in[19];
  const float* b1    = (const float*)d_in[20];
  const float* w2    = (const float*)d_in[21];
  const float* b2    = (const float*)d_in[22];

  char* ws = (char*)d_ws;
  size_t off = 0;
  auto take = [&](size_t bytes) {
    char* p = ws + off;
    off += (bytes + 255) & ~(size_t)255;
    return p;
  };
  float*  x    = (float*)take((size_t)SS * DD * 4);              // residual fp32
  __bf16* hbuf = (__bf16*)take((size_t)SS * DD * 2);             // rmsnorm out
  __bf16* qbuf = (__bf16*)take((size_t)SS * DD * 2);             // Q [S, H*64]
  __bf16* kbuf = (__bf16*)take((size_t)SS * (NKV * HD64) * 2);   // K [S, KV*64]
  __bf16* vtb  = (__bf16*)take((size_t)(NKV * HD64) * SS * 2);   // V^T [KV*64, S]
  __bf16* abuf = (__bf16*)take((size_t)SS * DD * 2);             // attn out
  __bf16* gbuf = (__bf16*)take((size_t)SS * FFD * 2);            // gate / gated
  __bf16* ubuf = (__bf16*)take((size_t)SS * FFD * 2);            // up
  __bf16* wbf  = (__bf16*)take((size_t)FFD * DD * 2);            // dequant weights

  auto dq = [&](const int* W, int n) {
    dequant_kernel<<<(n / 4 + 255) / 256, 256, 0, stream>>>(W, wbf, n);
  };

  embed_kernel<<<SS, 256, 0, stream>>>(ids, emb, x);

  int tQ  = (SS / 32) * (DD / 64);
  int tKV = (SS / 32) * ((NKV * HD64) / 64);
  int tF  = (SS / 32) * (FFD / 64);

  for (int l = 0; l < 4; ++l) {
    rms_kernel<<<SS, 256, 0, stream>>>(x, lnA + (size_t)l * DD, hbuf);

    dq(Wq + (size_t)l * DD * DD, DD * DD);
    gemm_bf16w<<<(tQ + 7) / 8, 256, 0, stream>>>(hbuf, wbf, sq + (size_t)l * DD,
                                                 SS, DD, DD, 1, nullptr, qbuf);
    dq(Wk + (size_t)l * (NKV * HD64) * DD, NKV * HD64 * DD);
    gemm_bf16w<<<(tKV + 7) / 8, 256, 0, stream>>>(hbuf, wbf, sk + (size_t)l * (NKV * HD64),
                                                  SS, NKV * HD64, DD, 1, nullptr, kbuf);
    dq(Wv + (size_t)l * (NKV * HD64) * DD, NKV * HD64 * DD);
    gemm_bf16w<<<(tKV + 7) / 8, 256, 0, stream>>>(hbuf, wbf, sv + (size_t)l * (NKV * HD64),
                                                  SS, NKV * HD64, DD, 2, nullptr, vtb);

    int nrope = SS * (NH + NKV) * (HD64 / 2);
    rope_kernel<<<(nrope + 255) / 256, 256, 0, stream>>>(qbuf, kbuf, nrope);

    attn_kernel<<<dim3(NH, SS / 16 / 8), 256, 0, stream>>>(qbuf, kbuf, vtb, abuf);

    dq(Wo + (size_t)l * DD * DD, DD * DD);
    gemm_bf16w<<<(tQ + 7) / 8, 256, 0, stream>>>(abuf, wbf, so + (size_t)l * DD,
                                                 SS, DD, DD, 0, x, nullptr);

    rms_kernel<<<SS, 256, 0, stream>>>(x, lnM + (size_t)l * DD, hbuf);

    dq(Wg + (size_t)l * FFD * DD, FFD * DD);
    gemm_bf16w<<<(tF + 7) / 8, 256, 0, stream>>>(hbuf, wbf, sg + (size_t)l * FFD,
                                                 SS, FFD, DD, 1, nullptr, gbuf);
    dq(Wu + (size_t)l * FFD * DD, FFD * DD);
    gemm_bf16w<<<(tF + 7) / 8, 256, 0, stream>>>(hbuf, wbf, su + (size_t)l * FFD,
                                                 SS, FFD, DD, 1, nullptr, ubuf);

    int nsm = SS * FFD;
    silumul_kernel<<<(nsm + 255) / 256, 256, 0, stream>>>(gbuf, ubuf, gbuf, nsm);

    dq(Wd + (size_t)l * DD * FFD, DD * FFD);
    gemm_bf16w<<<(tQ + 7) / 8, 256, 0, stream>>>(gbuf, wbf, sd + (size_t)l * DD,
                                                 SS, DD, FFD, 0, x, nullptr);
  }

  head_kernel<<<1, 512, 0, stream>>>(x, fnorm, w1, b1, w2, b2, (float*)d_out);
}
